// KPlexPool_6803228196878
// MI455X (gfx1250) — matrix operand order repaired and checked by reading, not compiled, not run on hardware
//
#include <hip/hip_runtime.h>
#include <math.h>

// ---------------------------------------------------------------------------
// Problem constants (from the reference)
// ---------------------------------------------------------------------------
#define NN   150000
#define EE   1500000
#define FIN  64
#define HH   64
#define CC   50000
#define PP   300000
#define EE2  600000
#define BB   16
#define NCLS 10

typedef __attribute__((ext_vector_type(16))) _Float16 v16h;
typedef __attribute__((ext_vector_type(8)))  float    v8f;

// ---------------------------------------------------------------------------
// WMMA GEMM:  out[:, ocol:ocol+64] = act(A[n,K] @ W[K,64] (+bias))
// One wave computes a 16x64 tile (4 WMMA 16x16 tiles, KSTEPS k-steps of 32).
// Block = 256 threads = 8 waves = 128 rows.
//
// W is staged into LDS PRE-SWIZZLED into this kernel's 16-bit B-matrix fragment
// layout (contiguous k order, same as A): fragment (ks,nt), lane L holds halves j=0..15 with
//   K = ks*32 + (L>=16 ? 16 : 0) + j,   N = nt*16 + (L&15)
// so each fragment is one contiguous 32-byte LDS read per lane.
// ---------------------------------------------------------------------------
template <int KSTEPS>
__global__ void __launch_bounds__(256)
gemm_n64_wmma(const float* __restrict__ A, int lda,
              const float* __restrict__ W,     // K x 64 row-major
              const float* __restrict__ bias,  // 64 (may be null)
              float* __restrict__ out, int ldo, int ocol,
              int nrows, int doRelu, int hasBias)
{
    constexpr int K = KSTEPS * 32;
    __shared__ alignas(32) _Float16 sW[KSTEPS * 4 * 32 * 16];  // == K*64

    const int tid = threadIdx.x;
    // ---- stage W swizzled: linear index -> (ksIdx, nt, lane, j) ----
    for (int i = tid; i < K * 64; i += 256) {
        const int j    = i & 15;
        const int ln   = (i >> 4) & 31;
        const int nt   = (i >> 9) & 3;
        const int ksI  = i >> 11;
        const int k    = ksI * 32 + ((ln & 16) ? 16 : 0) + j;
        const int n    = nt * 16 + (ln & 15);
        sW[i] = (_Float16)W[k * 64 + n];
    }
    __syncthreads();

    const int wave = tid >> 5;
    const int lane = tid & 31;
    const long tileRow = ((long)blockIdx.x * 8 + wave) * 16;
    if (tileRow >= nrows) return;           // wave-uniform exit: EXEC stays full

    const int hi  = (lane >= 16) ? 1 : 0;   // lane group
    const int lm  = lane & 15;
    long arow = tileRow + lm;               // A row this lane loads
    if (arow >= nrows) arow = nrows - 1;    // clamp (stores are predicated)

    const float* arowp = A + arow * (long)lda;
    const v16h*  bfrag = (const v16h*)sW;   // fragment (ksI*4+nt)*32 + lane

    v8f acc[4] = {};

#pragma unroll
    for (int ksI = 0; ksI < KSTEPS; ++ksI) {
        const int ks = ksI * 32;
        // ---- A fragment: two contiguous 8-float runs -> 4x b128 loads ----
        const float4* ap0 = (const float4*)(arowp + ks + hi * 16);
        const float4* ap1 = (const float4*)(arowp + ks + hi * 16 + 8);
        __builtin_prefetch(arowp + ks + 32, 0, 0);
        const float4 f0 = ap0[0], f1 = ap0[1];
        const float4 f2 = ap1[0], f3 = ap1[1];
        v16h a;
        a[0]  = (_Float16)f0.x; a[1]  = (_Float16)f0.y;
        a[2]  = (_Float16)f0.z; a[3]  = (_Float16)f0.w;
        a[4]  = (_Float16)f1.x; a[5]  = (_Float16)f1.y;
        a[6]  = (_Float16)f1.z; a[7]  = (_Float16)f1.w;
        a[8]  = (_Float16)f2.x; a[9]  = (_Float16)f2.y;
        a[10] = (_Float16)f2.z; a[11] = (_Float16)f2.w;
        a[12] = (_Float16)f3.x; a[13] = (_Float16)f3.y;
        a[14] = (_Float16)f3.z; a[15] = (_Float16)f3.w;

        // ---- 4 B fragments (one 32B LDS read each), then WMMA ----
#pragma unroll
        for (int nt = 0; nt < 4; ++nt) {
            const v16h b = bfrag[(ksI * 4 + nt) * 32 + lane];
            acc[nt] = __builtin_amdgcn_wmma_f32_16x16x32_f16(
                false, a, false, b, (short)0, acc[nt], false, false);
        }
    }

    // ---- store: VGPR r -> row tileRow + r + hi*8, col nt*16+lm ----
#pragma unroll
    for (int nt = 0; nt < 4; ++nt) {
        const int ncol = nt * 16 + lm;
        const float bi = hasBias ? bias[ncol] : 0.0f;
#pragma unroll
        for (int r = 0; r < 8; ++r) {
            const long orow = tileRow + r + hi * 8;
            if (orow < nrows) {
                float v = acc[nt][r] + bi;
                if (doRelu) v = fmaxf(v, 0.0f);
                out[orow * (long)ldo + ocol + ncol] = v;
            }
        }
    }
}

// ---------------------------------------------------------------------------
// Elementwise / graph kernels
// ---------------------------------------------------------------------------
__global__ void zero_f32(float* p, long n)
{
    long i = (long)blockIdx.x * blockDim.x + threadIdx.x;
    if (i < n) p[i] = 0.0f;
}

__global__ void deg_accum(const int* __restrict__ col, const float* __restrict__ w,
                          float* __restrict__ deg, long nE)
{
    long e = (long)blockIdx.x * blockDim.x + threadIdx.x;
    if (e < nE) atomicAdd(&deg[col[e]], w[e]);
}

// dis = rsqrt(deg + 1)  (self loop weight 1 => always > 0, matching the ref)
__global__ void deg_to_dis(float* d, long n)
{
    long i = (long)blockIdx.x * blockDim.x + threadIdx.x;
    if (i < n) d[i] = rsqrtf(d[i] + 1.0f);
}

// agg[i,:] = dis[i]^2 * h[i,:]  (self-loop contribution; full overwrite, f4-vec)
__global__ void selfloop_init(const float* __restrict__ dis, const float* __restrict__ h,
                              float* __restrict__ agg, long n)
{
    long i = (long)blockIdx.x * blockDim.x + threadIdx.x;   // over n*16 float4s
    if (i < n * 16) {
        const float d = dis[i >> 4];
        const float d2 = d * d;
        float4 v = ((const float4*)h)[i];
        v.x *= d2; v.y *= d2; v.z *= d2; v.w *= d2;
        ((float4*)agg)[i] = v;
    }
}

// wave per edge: agg[col,:] += dis[row]*w*dis[col] * h[row,:]
__global__ void __launch_bounds__(256)
edge_aggregate(const int* __restrict__ row, const int* __restrict__ col,
               const float* __restrict__ w, const float* __restrict__ dis,
               const float* __restrict__ h, float* __restrict__ agg, long nE)
{
    const int lane = threadIdx.x & 31;
    long e = (long)blockIdx.x * 8 + (threadIdx.x >> 5);
    if (e >= nE) return;
    const int r = row[e], c = col[e];
    const float nrm = dis[r] * w[e] * dis[c];
    const float* hr = h + (long)r * 64;
    float* ac = agg + (long)c * 64;
    atomicAdd(&ac[lane],      nrm * hr[lane]);
    atomicAdd(&ac[lane + 32], nrm * hr[lane + 32]);
}

// out[row, ocol + f] = relu(agg[row,f] + b[f])   (float4-vectorized)
__global__ void bias_relu_to(const float* __restrict__ agg, const float* __restrict__ b,
                             float* __restrict__ out, int ldo, int ocol, long n)
{
    long i = (long)blockIdx.x * blockDim.x + threadIdx.x;   // over n*16 float4s
    if (i < n * 16) {
        const long rrow = i >> 4;
        const int  f4   = (int)(i & 15);
        float4 v = ((const float4*)agg)[i];
        const float4 bb = ((const float4*)b)[f4];
        v.x = fmaxf(v.x + bb.x, 0.0f);
        v.y = fmaxf(v.y + bb.y, 0.0f);
        v.z = fmaxf(v.z + bb.z, 0.0f);
        v.w = fmaxf(v.w + bb.w, 0.0f);
        *(float4*)(out + rrow * (long)ldo + ocol + f4 * 4) = v;
    }
}

// batch pooling: xs[b, col0+f] += x[row,f];  xs[b, col0+64+f] = max(...)
// x entries are post-ReLU (>=0) so uint atomicMax with init 0 is exact,
// and empty segments stay 0 (matching _segmax0).
__global__ void __launch_bounds__(256)
pool_batch(const float* __restrict__ x, const int* __restrict__ batch,
           float* __restrict__ xs, int col0, long n)
{
    const int lane = threadIdx.x & 31;
    long rw = (long)blockIdx.x * 8 + (threadIdx.x >> 5);
    if (rw >= n) return;
    const int b = batch[rw];
    const float v0 = x[rw * 64 + lane];
    const float v1 = x[rw * 64 + lane + 32];
    float* xb = xs + (long)b * 256 + col0;
    atomicAdd(&xb[lane],      v0);
    atomicAdd(&xb[lane + 32], v1);
    atomicMax((unsigned int*)&xb[64 + lane],      __float_as_uint(v0));
    atomicMax((unsigned int*)&xb[64 + lane + 32], __float_as_uint(v1));
}

// cover pooling: xp[c, 0:64] += x[cover_n[p],:];  xp[c, 64:128] = max(...)
__global__ void __launch_bounds__(256)
pool_cover(const float* __restrict__ x, const int* __restrict__ cn,
           const int* __restrict__ cc, float* __restrict__ xp, long P)
{
    const int lane = threadIdx.x & 31;
    long p = (long)blockIdx.x * 8 + (threadIdx.x >> 5);
    if (p >= P) return;
    const int nidx = cn[p], cidx = cc[p];
    const float v0 = x[(long)nidx * 64 + lane];
    const float v1 = x[(long)nidx * 64 + lane + 32];
    float* dst = xp + (long)cidx * 128;
    atomicAdd(&dst[lane],      v0);
    atomicAdd(&dst[lane + 32], v1);
    atomicMax((unsigned int*)&dst[64 + lane],      __float_as_uint(v0));
    atomicMax((unsigned int*)&dst[64 + lane + 32], __float_as_uint(v1));
}

// ---------------------------------------------------------------------------
// Head: BN over B=16 rows of [16,256], lin1(256->64)+relu, lin2(64->10), softmax
// ---------------------------------------------------------------------------
__global__ void __launch_bounds__(256)
head_kernel(const float* __restrict__ xs,
            const float* __restrict__ gamma, const float* __restrict__ beta,
            const float* __restrict__ W1, const float* __restrict__ b1,
            const float* __restrict__ W2, const float* __restrict__ b2,
            float* __restrict__ out)
{
    __shared__ float hn[16 * 256];
    __shared__ float h1[16 * 64];
    __shared__ float lg[16 * NCLS];
    const int t = threadIdx.x;

    // BatchNorm: one thread per column
    {
        const int c = t;
        float mu = 0.0f;
        for (int r = 0; r < 16; ++r) mu += xs[r * 256 + c];
        mu *= (1.0f / 16.0f);
        float var = 0.0f;
        for (int r = 0; r < 16; ++r) { float d = xs[r * 256 + c] - mu; var += d * d; }
        var *= (1.0f / 16.0f);
        const float inv = rsqrtf(var + 1e-5f);
        const float g = gamma[c], be = beta[c];
        for (int r = 0; r < 16; ++r)
            hn[r * 256 + c] = (xs[r * 256 + c] - mu) * inv * g + be;
    }
    __syncthreads();

    // lin1 + relu
    for (int o = t; o < 16 * 64; o += 256) {
        const int r = o >> 6, c = o & 63;
        float s = b1[c];
        for (int k = 0; k < 256; ++k) s += hn[r * 256 + k] * W1[k * 64 + c];
        h1[o] = fmaxf(s, 0.0f);
    }
    __syncthreads();

    // lin2
    for (int o = t; o < 16 * NCLS; o += 256) {
        const int r = o / NCLS, c = o % NCLS;
        float s = b2[c];
        for (int k = 0; k < 64; ++k) s += h1[r * 64 + k] * W2[k * NCLS + c];
        lg[o] = s;
    }
    __syncthreads();

    // softmax per row
    if (t < 16) {
        float m = -3.0e38f;
        for (int c = 0; c < NCLS; ++c) m = fmaxf(m, lg[t * NCLS + c]);
        float e[NCLS], sum = 0.0f;
        for (int c = 0; c < NCLS; ++c) { e[c] = __expf(lg[t * NCLS + c] - m); sum += e[c]; }
        const float inv = 1.0f / sum;
        for (int c = 0; c < NCLS; ++c) out[t * NCLS + c] = e[c] * inv;
    }
}

// ---------------------------------------------------------------------------
// Host side
// ---------------------------------------------------------------------------
static inline void launch_gemm(const float* A, int lda, const float* W, const float* bias,
                               float* out, int ldo, int ocol, int n, int K,
                               int relu, int hasBias, hipStream_t s)
{
    const int blocks = (n + 127) / 128;
    if (K == 64)
        gemm_n64_wmma<2><<<blocks, 256, 0, s>>>(A, lda, W, bias, out, ldo, ocol, n, relu, hasBias);
    else
        gemm_n64_wmma<4><<<blocks, 256, 0, s>>>(A, lda, W, bias, out, ldo, ocol, n, relu, hasBias);
}

extern "C" void kernel_launch(void* const* d_in, const int* in_sizes, int n_in,
                              void* d_out, int out_size, void* d_ws, size_t ws_size,
                              hipStream_t stream)
{
    (void)in_sizes; (void)n_in; (void)out_size; (void)ws_size;

    // ---- inputs (setup_inputs order) ----
    const float* x       = (const float*)d_in[0];
    const int*   ei      = (const int*)  d_in[1];   // [2,E] : row=ei, col=ei+E
    const float* w       = (const float*)d_in[2];
    const int*   batch   = (const int*)  d_in[3];
    const int*   cover_n = (const int*)  d_in[4];
    const int*   cover_c = (const int*)  d_in[5];
    const int*   ei2     = (const int*)  d_in[6];
    const float* w2      = (const float*)d_in[7];
    const int*   batch2  = (const int*)  d_in[8];
    const float* W_in0 = (const float*)d_in[9],  *b_in0 = (const float*)d_in[10];
    const float* W_in1 = (const float*)d_in[11], *b_in1 = (const float*)d_in[12];
    const float* Wl_in = (const float*)d_in[13], *bl_in = (const float*)d_in[14];
    const float* W_b0  = (const float*)d_in[15], *b_b0  = (const float*)d_in[16];
    const float* W_b1  = (const float*)d_in[17], *b_b1  = (const float*)d_in[18];
    const float* Wl_b  = (const float*)d_in[19], *bl_b  = (const float*)d_in[20];
    const float* bn_g  = (const float*)d_in[21], *bn_b  = (const float*)d_in[22];
    const float* l1W   = (const float*)d_in[23], *l1b   = (const float*)d_in[24];
    const float* l2W   = (const float*)d_in[25], *l2b   = (const float*)d_in[26];

    // ---- workspace layout ----
    char* ws = (char*)d_ws;
    size_t off = 0;
    auto take = [&](size_t bytes) { char* p = ws + off; off += (bytes + 255) & ~(size_t)255; return p; };
    float* dis = (float*)take((size_t)NN * 4);        // also deg scratch (stage C reuses)
    float* h   = (float*)take((size_t)NN * 64 * 4);   // GEMM out; also block output (aliased in time)
    float* agg = (float*)take((size_t)NN * 64 * 4);
    float* cat = (float*)take((size_t)NN * 128 * 4);  // [x1 | x2] JK-cat buffer
    float* xp  = (float*)take((size_t)CC * 128 * 4);  // pooled features [sum | max]
    float* xs  = (float*)take((size_t)BB * 256 * 4);  // readout rows

    auto g1 = [](long n) { return (unsigned)((n + 255) / 256); };

    // ================= Stage A: block on the big graph =================
    zero_f32<<<g1(NN), 256, 0, stream>>>(dis, NN);
    deg_accum<<<g1(EE), 256, 0, stream>>>(ei + EE, w, dis, EE);
    deg_to_dis<<<g1(NN), 256, 0, stream>>>(dis, NN);

    // GCN layer 1: h = x @ W_in0 ; aggregate ; x1 = relu(agg + b) -> cat[:,0:64]
    launch_gemm(x, 64, W_in0, nullptr, h, 64, 0, NN, 64, 0, 0, stream);
    selfloop_init<<<g1((long)NN * 16), 256, 0, stream>>>(dis, h, agg, NN);
    edge_aggregate<<<(unsigned)((EE + 7) / 8), 256, 0, stream>>>(ei, ei + EE, w, dis, h, agg, EE);
    bias_relu_to<<<g1((long)NN * 16), 256, 0, stream>>>(agg, b_in0, cat, 128, 0, NN);

    // GCN layer 2: h = x1 @ W_in1 ; aggregate ; x2 = relu(agg + b) -> cat[:,64:128]
    launch_gemm(cat, 128, W_in1, nullptr, h, 64, 0, NN, 64, 0, 0, stream);
    selfloop_init<<<g1((long)NN * 16), 256, 0, stream>>>(dis, h, agg, NN);
    edge_aggregate<<<(unsigned)((EE + 7) / 8), 256, 0, stream>>>(ei, ei + EE, w, dis, h, agg, EE);
    bias_relu_to<<<g1((long)NN * 16), 256, 0, stream>>>(agg, b_in1, cat, 128, 64, NN);

    // JK-cat linear + outer relu: x1big = relu(cat @ Wl_in + bl_in) -> h (reused)
    launch_gemm(cat, 128, Wl_in, bl_in, h, 64, 0, NN, 128, 1, 1, stream);

    // ================= Stage B: pooling =================
    zero_f32<<<g1((long)BB * 256), 256, 0, stream>>>(xs, (long)BB * 256);
    pool_batch<<<(unsigned)((NN + 7) / 8), 256, 0, stream>>>(h, batch, xs, 0, NN);
    zero_f32<<<g1((long)CC * 128), 256, 0, stream>>>(xp, (long)CC * 128);
    pool_cover<<<(unsigned)((PP + 7) / 8), 256, 0, stream>>>(h, cover_n, cover_c, xp, PP);

    // ================= Stage C: block on the pooled graph =================
    zero_f32<<<g1(CC), 256, 0, stream>>>(dis, CC);
    deg_accum<<<g1(EE2), 256, 0, stream>>>(ei2 + EE2, w2, dis, EE2);
    deg_to_dis<<<g1(CC), 256, 0, stream>>>(dis, CC);

    launch_gemm(xp, 128, W_b0, nullptr, h, 64, 0, CC, 128, 0, 0, stream);
    selfloop_init<<<g1((long)CC * 16), 256, 0, stream>>>(dis, h, agg, CC);
    edge_aggregate<<<(unsigned)((EE2 + 7) / 8), 256, 0, stream>>>(ei2, ei2 + EE2, w2, dis, h, agg, EE2);
    bias_relu_to<<<g1((long)CC * 16), 256, 0, stream>>>(agg, b_b0, cat, 128, 0, CC);

    launch_gemm(cat, 128, W_b1, nullptr, h, 64, 0, CC, 64, 0, 0, stream);
    selfloop_init<<<g1((long)CC * 16), 256, 0, stream>>>(dis, h, agg, CC);
    edge_aggregate<<<(unsigned)((EE2 + 7) / 8), 256, 0, stream>>>(ei2, ei2 + EE2, w2, dis, h, agg, EE2);
    bias_relu_to<<<g1((long)CC * 16), 256, 0, stream>>>(agg, b_b1, cat, 128, 64, CC);

    launch_gemm(cat, 128, Wl_b, bl_b, h, 64, 0, CC, 128, 1, 1, stream);
    pool_batch<<<(unsigned)((CC + 7) / 8), 256, 0, stream>>>(h, batch2, xs, 128, CC);

    // ================= Stage D: head =================
    head_kernel<<<1, 256, 0, stream>>>(xs, bn_g, bn_b, l1W, l1b, l2W, l2b, (float*)d_out);
}
